// MultiheadPopulationAttention_NotResidual_59682865545948
// MI455X (gfx1250) — matrix operand order repaired and checked by reading, not compiled, not run on hardware
//
#include <hip/hip_runtime.h>
#include <hip/hip_bf16.h>

#define BB 2
#define LL 2048
#define EE 1024
#define HH 16
#define DH 64
#define HID 4096
#define INV_SCALE 0.125f   // 1/sqrt(DH)

#ifndef USE_ASYNC
#define USE_ASYNC 1
#endif

typedef _Float16 v16h __attribute__((ext_vector_type(16)));
typedef _Float16 v8h  __attribute__((ext_vector_type(8)));
typedef _Float16 v4h  __attribute__((ext_vector_type(4)));
typedef float    v8f  __attribute__((ext_vector_type(8)));

__device__ __forceinline__ v16h cat8(v8h a, v8h b) {
    return __builtin_shufflevector(a, b, 0,1,2,3,4,5,6,7,8,9,10,11,12,13,14,15);
}

__device__ __forceinline__ v8f wmma_f16(v16h a, v16h b, v8f c) {
    // D = A(16x32 f16) * B(32x16 f16) + C(16x16 f32)
    return __builtin_amdgcn_wmma_f32_16x16x32_f16(false, a, false, b, (short)0, c, false, false);
}

// Async copy of 16 bytes global -> LDS via the CDNA5 async-DMA path (ASYNCcnt).
__device__ __forceinline__ void lds_copy16_async(_Float16* l, const _Float16* g) {
#if USE_ASYNC
    unsigned loff = (unsigned)(uintptr_t)(void*)l;          // low 32 bits = LDS offset
    unsigned long long ga = (unsigned long long)(uintptr_t)g;
    asm volatile("global_load_async_to_lds_b128 %0, %1, off"
                 :: "v"(loff), "v"(ga) : "memory");
#else
    *(v8h*)l = *(const v8h*)g;
#endif
}
__device__ __forceinline__ void lds_async_wait() {
#if USE_ASYNC
    asm volatile("s_wait_asynccnt 0x0" ::: "memory");
#endif
}

// ---------------------------------------------------------------- converts
__global__ void k_f32_to_f16(const float* __restrict__ in, _Float16* __restrict__ out, int n) {
    int i = blockIdx.x * blockDim.x + threadIdx.x;
    if (i < n) out[i] = (_Float16)in[i];
}

// ---------------------------------------------------------------- GEMM: C = A[MxK] @ W[KxN] + bias, optional exact GELU, f16 out
// Block tile 128x64 (8 waves, each 32x32 = 2x2 WMMA tiles), k-step 64.
template<bool GELU>
__global__ void k_gemm(const _Float16* __restrict__ A, const _Float16* __restrict__ W,
                       const float* __restrict__ bias, _Float16* __restrict__ C,
                       int M, int N, int K) {
    __shared__ __align__(16) _Float16 sA[128 * 64];   // [m][k]
    __shared__ __align__(16) _Float16 sBt[64 * 64];   // [n][k] (transposed)
    const int tid  = threadIdx.x;
    const int lane = tid & 31;
    const int wave = tid >> 5;
    const int n0 = blockIdx.x * 64;
    const int m0 = blockIdx.y * 128;
    const int msub = (wave >> 1) * 32;
    const int nsub = (wave & 1) * 32;
    const int base0 = (lane >> 4) * 8;   // A-fragment K chunk base
    const int kbB   = (lane >> 4) * 8;   // B-fragment K chunk base

    v8f zf = {};
    v8f acc[2][2] = {{zf, zf}, {zf, zf}};

    for (int k0 = 0; k0 < K; k0 += 64) {
        // stage A tile 128x64 through the async-DMA path
        #pragma unroll
        for (int p = 0; p < 4; ++p) {
            int c = tid + p * 256;
            int row = c >> 3, cg = c & 7;
            lds_copy16_async(sA + row * 64 + cg * 8,
                             A + (size_t)(m0 + row) * K + k0 + cg * 8);
        }
        // stage W tile 64x64 transposed -> sBt[n][k]
        #pragma unroll
        for (int p = 0; p < 2; ++p) {
            int c = tid + p * 256;
            int kk = c >> 3, ng = c & 7;
            v8h v = *(const v8h*)(W + (size_t)(k0 + kk) * N + n0 + ng * 8);
            #pragma unroll
            for (int q = 0; q < 8; ++q) sBt[(ng * 8 + q) * 64 + kk] = v[q];
        }
        lds_async_wait();
        __syncthreads();
        #pragma unroll
        for (int kc = 0; kc < 2; ++kc) {
            v16h a[2], b[2];
            #pragma unroll
            for (int mt = 0; mt < 2; ++mt) {
                const _Float16* p = sA + (msub + mt * 16 + (lane & 15)) * 64 + kc * 32 + base0;
                a[mt] = cat8(*(const v8h*)p, *(const v8h*)(p + 16));
            }
            #pragma unroll
            for (int nt = 0; nt < 2; ++nt) {
                const _Float16* p = sBt + (nsub + nt * 16 + (lane & 15)) * 64 + kc * 32 + kbB;
                b[nt] = cat8(*(const v8h*)p, *(const v8h*)(p + 16));
            }
            #pragma unroll
            for (int mt = 0; mt < 2; ++mt)
                #pragma unroll
                for (int nt = 0; nt < 2; ++nt)
                    acc[mt][nt] = wmma_f16(a[mt], b[nt], acc[mt][nt]);
        }
        __syncthreads();
    }
    #pragma unroll
    for (int nt = 0; nt < 2; ++nt) {
        const int n = n0 + nsub + nt * 16 + (lane & 15);
        const float bn = bias[n];
        #pragma unroll
        for (int mt = 0; mt < 2; ++mt) {
            const int mrow = m0 + msub + mt * 16 + 8 * (lane >> 4);
            #pragma unroll
            for (int r = 0; r < 8; ++r) {
                float v = acc[mt][nt][r] + bn;
                if (GELU) v = 0.5f * v * (1.0f + erff(v * 0.70710678118f));
                C[(size_t)(mrow + r) * N + n] = (_Float16)v;
            }
        }
    }
}

// ---------------------------------------------------------------- pass A: per-(b,h,j) stats m_j, c_j = x_j / den_j
__global__ void k_attn_stats(const _Float16* __restrict__ Q, const _Float16* __restrict__ Kmat,
                             const float* __restrict__ x,
                             float* __restrict__ mOut, float* __restrict__ cOut) {
    const int jblk = blockIdx.x * 64;
    const int h = blockIdx.y, b = blockIdx.z;
    const int tid = threadIdx.x, lane = tid & 31, wave = tid >> 5;
    const int jt = wave & 3;       // which 16-j tile
    const int ih = wave >> 2;      // which half of the i range
    const _Float16* Qh = Q    + (size_t)b * LL * EE + (size_t)h * DH;
    const _Float16* Kh = Kmat + (size_t)b * LL * EE + (size_t)h * DH;
    const float* xb = x + (size_t)b * LL;
    const int base0 = (lane >> 4) * 8;
    const int kbB   = (lane >> 4) * 8;

    // preload K B-fragments (fixed j's for this wave), d split 0..31 / 32..63
    const int j = jblk + jt * 16 + (lane & 15);
    v16h bk0 = cat8(*(const v8h*)(Kh + (size_t)j * EE + kbB),
                    *(const v8h*)(Kh + (size_t)j * EE + kbB + 16));
    v16h bk1 = cat8(*(const v8h*)(Kh + (size_t)j * EE + 32 + kbB),
                    *(const v8h*)(Kh + (size_t)j * EE + 32 + kbB + 16));

    float m = -3.0e38f, ssum = 0.0f;
    for (int it = 0; it < 64; ++it) {
        const int i0 = ih * (LL / 2) + it * 16;
        const int irow = i0 + (lane & 15);
        v16h aq0 = cat8(*(const v8h*)(Qh + (size_t)irow * EE + base0),
                        *(const v8h*)(Qh + (size_t)irow * EE + base0 + 16));
        v16h aq1 = cat8(*(const v8h*)(Qh + (size_t)irow * EE + 32 + base0),
                        *(const v8h*)(Qh + (size_t)irow * EE + 32 + base0 + 16));
        v8f c = {};
        c = wmma_f16(aq0, bk0, c);
        c = wmma_f16(aq1, bk1, c);
        // element (r, lane): i = i0 + r + 8*(lane>>4), j = column (lane&15)
        float sv[8];
        float tm = -3.0e38f;
        #pragma unroll
        for (int r = 0; r < 8; ++r) { sv[r] = c[r] * INV_SCALE; tm = fmaxf(tm, sv[r]); }
        tm = fmaxf(tm, __shfl_xor(tm, 16));      // combine M halves (same j)
        const float nm = fmaxf(m, tm);
        float part = 0.0f;
        const int ibase = i0 + 8 * (lane >> 4);
        #pragma unroll
        for (int r = 0; r < 8; ++r) part += xb[ibase + r] * __expf(sv[r] - nm);
        part += __shfl_xor(part, 16);
        ssum = ssum * __expf(m - nm) + part;
        m = nm;
    }
    __shared__ float sm[8][16], ss[8][16];
    if ((lane >> 4) == 0) { sm[wave][lane & 15] = m; ss[wave][lane & 15] = ssum; }
    __syncthreads();
    if (tid < 64) {
        const int w0 = tid >> 4, idx = tid & 15;
        const float m0v = sm[w0][idx],     m1v = sm[w0 + 4][idx];
        const float s0  = ss[w0][idx],     s1  = ss[w0 + 4][idx];
        const float M = fmaxf(m0v, m1v);
        const float den = s0 * __expf(m0v - M) + s1 * __expf(m1v - M);
        const int jj = jblk + tid;
        const size_t o = ((size_t)b * HH + h) * LL + jj;
        mOut[o] = M;
        cOut[o] = xb[jj] / den;
    }
}

// ---------------------------------------------------------------- pass B: O[i,d] = x_i * sum_j exp(S-m_j)*c_j * V[j,d]
__global__ void k_attn_apply(const _Float16* __restrict__ Q, const _Float16* __restrict__ Kmat,
                             const _Float16* __restrict__ V, const float* __restrict__ x,
                             const float* __restrict__ mIn, const float* __restrict__ cIn,
                             _Float16* __restrict__ O) {
    const int h = blockIdx.y, b = blockIdx.z;
    const int tid = threadIdx.x, lane = tid & 31, wave = tid >> 5;
    const int i0 = blockIdx.x * 128 + wave * 16;
    __shared__ __align__(16) _Float16 sVt[DH * 32];      // [d][j] transposed V tile
    __shared__ __align__(16) _Float16 sP[8][16 * 32];    // per-wave P tile [i][j]
    const _Float16* Qh = Q    + (size_t)b * LL * EE + (size_t)h * DH;
    const _Float16* Kh = Kmat + (size_t)b * LL * EE + (size_t)h * DH;
    const _Float16* Vh = V    + (size_t)b * LL * EE + (size_t)h * DH;
    const float* mrow = mIn + ((size_t)b * HH + h) * LL;
    const float* crow = cIn + ((size_t)b * HH + h) * LL;
    const int base0 = (lane >> 4) * 8;
    const int kbB   = (lane >> 4) * 8;

    // preload Q A-fragments for this wave's i-tile
    const int irow = i0 + (lane & 15);
    v16h aq0 = cat8(*(const v8h*)(Qh + (size_t)irow * EE + base0),
                    *(const v8h*)(Qh + (size_t)irow * EE + base0 + 16));
    v16h aq1 = cat8(*(const v8h*)(Qh + (size_t)irow * EE + 32 + base0),
                    *(const v8h*)(Qh + (size_t)irow * EE + 32 + base0 + 16));

    v8f zf = {};
    v8f acc[4] = {zf, zf, zf, zf};

    for (int j0 = 0; j0 < LL; j0 += 32) {
        // stage V^T: sVt[d][j], 32 j x 64 d
        {
            const int jl = tid & 31, dg = tid >> 5;
            if (j0 + 32 < LL) __builtin_prefetch(Vh + (size_t)(j0 + 32 + jl) * EE, 0, 1);
            v8h v = *(const v8h*)(Vh + (size_t)(j0 + jl) * EE + dg * 8);
            #pragma unroll
            for (int q = 0; q < 8; ++q) sVt[(dg * 8 + q) * 32 + jl] = v[q];
        }
        __syncthreads();
        // S tiles (2 x 16 j) -> P (f16, in LDS, A-layout source)
        #pragma unroll
        for (int jt = 0; jt < 2; ++jt) {
            const int j = j0 + jt * 16 + (lane & 15);
            v16h bk0 = cat8(*(const v8h*)(Kh + (size_t)j * EE + kbB),
                            *(const v8h*)(Kh + (size_t)j * EE + kbB + 16));
            v16h bk1 = cat8(*(const v8h*)(Kh + (size_t)j * EE + 32 + kbB),
                            *(const v8h*)(Kh + (size_t)j * EE + 32 + kbB + 16));
            v8f c = {};
            c = wmma_f16(aq0, bk0, c);
            c = wmma_f16(aq1, bk1, c);
            const float mj = mrow[j], cj = crow[j];
            #pragma unroll
            for (int r = 0; r < 8; ++r) {
                float p = __expf(c[r] * INV_SCALE - mj) * cj;
                sP[wave][(r + 8 * (lane >> 4)) * 32 + jt * 16 + (lane & 15)] = (_Float16)p;
            }
        }
        // read P back as A-fragment (16 i x 32 j)
        v16h aP = cat8(*(const v8h*)(&sP[wave][(lane & 15) * 32 + base0]),
                       *(const v8h*)(&sP[wave][(lane & 15) * 32 + base0 + 16]));
        #pragma unroll
        for (int dt = 0; dt < 4; ++dt) {
            v16h bV = cat8(*(const v8h*)(sVt + (dt * 16 + (lane & 15)) * 32 + kbB),
                           *(const v8h*)(sVt + (dt * 16 + (lane & 15)) * 32 + kbB + 16));
            acc[dt] = wmma_f16(aP, bV, acc[dt]);
        }
        __syncthreads();
    }
    const float* xb = x + (size_t)b * LL;
    const int ibase = i0 + 8 * (lane >> 4);
    #pragma unroll
    for (int dt = 0; dt < 4; ++dt) {
        const int d = h * DH + dt * 16 + (lane & 15);
        #pragma unroll
        for (int r = 0; r < 8; ++r) {
            float o = acc[dt][r] * xb[ibase + r];
            O[(size_t)(b * LL + ibase + r) * EE + d] = (_Float16)o;
        }
    }
}

// ---------------------------------------------------------------- final GEMV: y = Y1 @ W2 + b2
__global__ void k_final(const _Float16* __restrict__ Y1, const float* __restrict__ W2,
                        const float* __restrict__ b2, float* __restrict__ out) {
    const int row = blockIdx.x;
    const int tid = threadIdx.x;
    float p = 0.0f;
    for (int k = tid; k < HID; k += 256) p += (float)Y1[(size_t)row * HID + k] * W2[k];
    __shared__ float red[256];
    red[tid] = p;
    __syncthreads();
    for (int s = 128; s > 0; s >>= 1) {
        if (tid < s) red[tid] += red[tid + s];
        __syncthreads();
    }
    if (tid == 0) out[row] = red[0] + b2[0];
}

extern "C" void kernel_launch(void* const* d_in, const int* in_sizes, int n_in,
                              void* d_out, int out_size, void* d_ws, size_t ws_size,
                              hipStream_t stream) {
    const float* x  = (const float*)d_in[0];
    const float* z  = (const float*)d_in[1];
    const float* Wq = (const float*)d_in[2];
    const float* bq = (const float*)d_in[3];
    const float* Wk = (const float*)d_in[4];
    const float* bk = (const float*)d_in[5];
    const float* Wv = (const float*)d_in[6];
    const float* bv = (const float*)d_in[7];
    const float* W1 = (const float*)d_in[8];
    const float* b1 = (const float*)d_in[9];
    const float* W2 = (const float*)d_in[10];
    const float* b2 = (const float*)d_in[11];
    float* out = (float*)d_out;

    char* w = (char*)d_ws;
    size_t off = 0;
    auto alloc = [&](size_t bytes) -> void* {
        void* p = w + off;
        off = (off + bytes + 255) & ~(size_t)255;
        return p;
    };
    const int M = BB * LL;                                // 4096 rows
    _Float16* zh  = (_Float16*)alloc((size_t)M * EE * 2);
    _Float16* Wqh = (_Float16*)alloc((size_t)EE * EE * 2);
    _Float16* Wkh = (_Float16*)alloc((size_t)EE * EE * 2);
    _Float16* Wvh = (_Float16*)alloc((size_t)EE * EE * 2);
    _Float16* W1h = (_Float16*)alloc((size_t)EE * HID * 2);
    _Float16* Qb  = (_Float16*)alloc((size_t)M * EE * 2);
    _Float16* Kb  = (_Float16*)alloc((size_t)M * EE * 2);
    _Float16* Vb  = (_Float16*)alloc((size_t)M * EE * 2);
    _Float16* Ob  = (_Float16*)alloc((size_t)M * EE * 2);
    _Float16* Y1  = (_Float16*)alloc((size_t)M * HID * 2);
    float*    mB  = (float*)alloc((size_t)BB * HH * LL * 4);
    float*    cB  = (float*)alloc((size_t)BB * HH * LL * 4);

    // f32 -> f16 converts
    auto cvt = [&](const float* src, _Float16* dst, int n) {
        k_f32_to_f16<<<(n + 255) / 256, 256, 0, stream>>>(src, dst, n);
    };
    cvt(z,  zh,  M * EE);
    cvt(Wq, Wqh, EE * EE);
    cvt(Wk, Wkh, EE * EE);
    cvt(Wv, Wvh, EE * EE);
    cvt(W1, W1h, EE * HID);

    // QKV projections: [4096x1024] @ [1024x1024] + bias
    dim3 gQKV(EE / 64, M / 128);
    k_gemm<false><<<gQKV, 256, 0, stream>>>(zh, Wqh, bq, Qb, M, EE, EE);
    k_gemm<false><<<gQKV, 256, 0, stream>>>(zh, Wkh, bk, Kb, M, EE, EE);
    k_gemm<false><<<gQKV, 256, 0, stream>>>(zh, Wvh, bv, Vb, M, EE, EE);

    // attention pass A: column (query-axis) softmax stats
    dim3 gStats(LL / 64, HH, BB);
    k_attn_stats<<<gStats, 256, 0, stream>>>(Qb, Kb, x, mB, cB);

    // attention pass B: apply
    dim3 gApply(LL / 128, HH, BB);
    k_attn_apply<<<gApply, 256, 0, stream>>>(Qb, Kb, Vb, x, mB, cB, Ob);

    // FFN1 with exact GELU: [4096x1024] @ [1024x4096] + b1
    dim3 gFFN(HID / 64, M / 128);
    k_gemm<true><<<gFFN, 256, 0, stream>>>(Ob, W1h, b1, Y1, M, HID, EE);

    // final GEMV: [4096x4096] @ [4096x1] + b2
    k_final<<<M, 256, 0, stream>>>(Y1, W2, b2, out);
}
